// FBSNN_2267742732953
// MI455X (gfx1250) — compile-verified, matrix operand order checked by reading, not run on hardware
//
#include <hip/hip_runtime.h>

#define DT_C     0.01f
#define SIGMA_C  0.5f
#define NSTEPS_C 50

typedef __attribute__((ext_vector_type(16))) _Float16 v16h;
typedef __attribute__((ext_vector_type(8)))  _Float16 v8h;
typedef __attribute__((ext_vector_type(8)))  float    v8f;
typedef __attribute__((ext_vector_type(4)))  float    v4f;

// ---------------- cross-lane helpers (wave32) ----------------
// ds_swizzle group-of-32: offset = {xor[4:0]<<10 | or<<5 | and[4:0]}
template<int XM>
__device__ __forceinline__ float swzadd(float v) {
  int r = __builtin_amdgcn_ds_swizzle(__float_as_int(v), (XM << 10) | 0x1F);
  return v + __int_as_float(r);
}

// Fold 8 per-row partial sums (row = r + 8*half, partial over 4 columns/lane)
// across the 16 lanes of each half while splitting rows over lanes.
// Result: lane l holds the total of row  4*b3 + 2*b2 + 1*b1 + 8*(l>>4)
// (b_i = bit i of l). Lanes l and l^1 duplicate.
__device__ __forceinline__ float fold_rows(const float* v, int lane) {
  float a0 = swzadd<8>(v[0]);
  float a1 = swzadd<8>(v[1]);
  float a2 = swzadd<8>(v[2]);
  float a3 = swzadd<8>(v[3]);
  float a4 = swzadd<8>(v[4]);
  float a5 = swzadd<8>(v[5]);
  float a6 = swzadd<8>(v[6]);
  float a7 = swzadd<8>(v[7]);
  bool s3 = (lane & 8) != 0;
  float b0 = s3 ? a4 : a0;
  float b1 = s3 ? a5 : a1;
  float b2 = s3 ? a6 : a2;
  float b3 = s3 ? a7 : a3;
  b0 = swzadd<4>(b0); b1 = swzadd<4>(b1);
  b2 = swzadd<4>(b2); b3 = swzadd<4>(b3);
  bool s2 = (lane & 4) != 0;
  float c0 = s2 ? b2 : b0;
  float c1 = s2 ? b3 : b1;
  c0 = swzadd<2>(c0); c1 = swzadd<2>(c1);
  float d = (lane & 2) ? c1 : c0;
  return swzadd<1>(d);
}
__device__ __forceinline__ int fold_row_of(int lane) {
  return (((lane >> 3) & 1) << 2) | (((lane >> 2) & 1) << 1) |
         ((lane >> 1) & 1) | ((lane >> 4) << 3);
}

// ---------------- WMMA fragment helpers ----------------
// B fragment (32x16 fp16) for output tile tt (cols 16*tt..) and K-chunk c,
// loaded from LDS weight matrix stored transposed: wT[n][k], row stride 64.
__device__ __forceinline__ v16h bfrag_load(const _Float16* wT, int tt, int c, int lane) {
  int n  = 16 * tt + (lane & 15);
  int kb = 32 * c + ((lane >> 4) << 3);
  const _Float16* row = wT + n * 64;
  v8h lo = *(const v8h*)(row + kb);        // K = kb .. kb+7
  v8h hi = *(const v8h*)(row + kb + 16);   // K = kb+16 .. kb+23
  return __builtin_shufflevector(lo, hi, 0,1,2,3,4,5,6,7,8,9,10,11,12,13,14,15);
}

// Layer 1 straight into A-fragment layout (16x32 fp16 per chunk):
// element e of lane L holds column j = 32c + 8*(L>=16) + e + (e&8).
// a1[j] = t*w1[0][j] + y*w1[1][j] + b1[j];  h = relu(a1);
// tangent (input tangent = d/dy): dh = (a1>0) ? w1[1][j] : 0.
template<bool WANTD>
__device__ __forceinline__ void layer1_frags(float t, float y, const v4f* pk1,
                                             int lane, v16h h[2], v16h dh[2]) {
  int kb0 = (lane >> 4) << 3;
#pragma unroll
  for (int c = 0; c < 2; ++c) {
    v16h hh; v16h dd;
#pragma unroll
    for (int e = 0; e < 16; ++e) {
      int j = 32 * c + kb0 + e + (e & 8);
      v4f w = pk1[j];                       // {w1_t, w1_y, b1, 0}
      float a  = fmaf(t, w.x, fmaf(y, w.y, w.z));
      bool  on = a > 0.f;
      hh[e] = (_Float16)(on ? a : 0.f);
      if (WANTD) dd[e] = (_Float16)(on ? w.y : 0.f);
    }
    h[c] = hh;
    if (WANTD) dh[c] = dd;
  }
}

// Full MLP eval for a 16-sample tile. If WANTD, also forward-mode JVP wrt y.
// Returns per-lane scalars for sample m = lane&15 (duplicated across halves).
template<bool WANTD>
__device__ __forceinline__ void net_eval(
    float t, float y,
    const v4f* pk1, const _Float16* wT, const float* b2v, const float* w3v,
    float b3, int lane, float* stage /* [32] per-wave */,
    float* uOut, float* dOut) {
  int m = lane & 15;
  v16h hf[2], df[2];
  layer1_frags<WANTD>(t, y, pk1, lane, hf, df);

  float pu[8], pd[8];
#pragma unroll
  for (int r = 0; r < 8; ++r) { pu[r] = 0.f; pd[r] = 0.f; }

#pragma unroll
  for (int tt = 0; tt < 4; ++tt) {
    float bb = b2v[16 * tt + m];
    v8f aU = { bb, bb, bb, bb, bb, bb, bb, bb };
    v8f aD = {};
#pragma unroll
    for (int c = 0; c < 2; ++c) {
      v16h bf = bfrag_load(wT, tt, c, lane);
      aU = __builtin_amdgcn_wmma_f32_16x16x32_f16(false, hf[c], false, bf,
                                                  (short)0, aU, false, false);
      if (WANTD)
        aD = __builtin_amdgcn_wmma_f32_16x16x32_f16(false, df[c], false, bf,
                                                    (short)0, aD, false, false);
    }
    // layer 3 partial dot on C layout: element (row = r + 8*half, col = 16tt+m)
    float w3c = w3v[16 * tt + m];
#pragma unroll
    for (int r = 0; r < 8; ++r) {
      float a2 = aU[r];
      bool  on = a2 > 0.f;
      pu[r] = fmaf(on ? a2 : 0.f, w3c, pu[r]);
      if (WANTD) pd[r] = fmaf(on ? aD[r] : 0.f, w3c, pd[r]);
    }
  }
  float uf = fold_rows(pu, lane);
  int  row = fold_row_of(lane);
  if ((lane & 1) == 0) stage[row] = uf;          // even lanes cover rows 0..15
  if (WANTD) {
    float dv = fold_rows(pd, lane);
    if ((lane & 1) == 0) stage[16 + row] = dv;
  }
  __builtin_amdgcn_wave_barrier();               // keep LDS store->load ordered
  *uOut = stage[m] + b3;
  if (WANTD) *dOut = stage[16 + m];
}

// ---------------- main kernel ----------------
__global__ __launch_bounds__(256) void fbsnn_main(
    const float* __restrict__ dW, const float* __restrict__ y0_init,
    const float* __restrict__ Yw1, const float* __restrict__ Yb1,
    const float* __restrict__ Yw2, const float* __restrict__ Yb2,
    const float* __restrict__ Yw3, const float* __restrict__ Yb3,
    const float* __restrict__ Qw1, const float* __restrict__ Qb1,
    const float* __restrict__ Qw2, const float* __restrict__ Qb2,
    const float* __restrict__ Qw3, const float* __restrict__ Qb3,
    float* __restrict__ partials, int Btot) {
  __shared__ __attribute__((aligned(16))) _Float16 sYwT[64 * 64];
  __shared__ __attribute__((aligned(16))) _Float16 sQwT[64 * 64];
  __shared__ v4f   sYpk1[64];
  __shared__ v4f   sQpk1[64];
  __shared__ float sYb2[64], sYw3[64], sQb2[64], sQw3[64];
  __shared__ float sStage[8][32];
  __shared__ float sPart[8];

  int tid = threadIdx.x;
  // Stage weights once: w2 transposed to [n][k] fp16 for B-fragment loads.
  for (int i = tid; i < 64 * 64; i += 256) {
    int k = i >> 6, nn = i & 63;                 // i = k*64 + nn (row-major w2)
    sYwT[nn * 64 + k] = (_Float16)Yw2[i];
    sQwT[nn * 64 + k] = (_Float16)Qw2[i];
  }
  if (tid < 64) {
    sYpk1[tid] = (v4f){ Yw1[tid], Yw1[64 + tid], Yb1[tid], 0.f };
    sQpk1[tid] = (v4f){ Qw1[tid], Qw1[64 + tid], Qb1[tid], 0.f };
    sYb2[tid] = Yb2[tid];  sYw3[tid] = Yw3[tid];
    sQb2[tid] = Qb2[tid];  sQw3[tid] = Qw3[tid];
  }
  __syncthreads();

  int wave = tid >> 5;
  int lane = tid & 31;
  int m    = lane & 15;
  size_t gbase = ((size_t)blockIdx.x * 8 + wave) * 16;
  float* stage = sStage[wave];

  float yb3 = Yb3[0];
  float qb3 = Qb3[0];

  float y = y0_init[0];
  float Y0, dY0;
  net_eval<true>(0.f, y, sYpk1, sYwT, sYb2, sYw3, yb3, lane, stage, &Y0, &dY0);

  float acc = 0.f;
#pragma unroll 1
  for (int n = 0; n < NSTEPS_C; ++n) {
    float t0 = (float)n * DT_C;
    // control q = Qnet(t0, y)
    float q;
    net_eval<false>(t0, y, sQpk1, sQwT, sQb2, sQw3, qb3, lane, stage, &q, nullptr);

    float dwv = dW[(size_t)n * (size_t)Btot + gbase + m];
    if (n + 1 < NSTEPS_C)
      __builtin_prefetch(&dW[(size_t)(n + 1) * (size_t)Btot + gbase + m], 0, 0);

    float y1  = fmaf(q, DT_C, fmaf(SIGMA_C, dwv, y));          // Euler step
    float Y1t = Y0 - 0.5f * q * q * DT_C + SIGMA_C * dY0 * dwv; // Y0 - f*dt + z0*dW

    float Y1, dY1;
    net_eval<true>(t0 + DT_C, y1, sYpk1, sYwT, sYb2, sYw3, yb3, lane, stage,
                   &Y1, &dY1);

    float resid = Y1 - Y1t;
    if (lane < 16) acc = fmaf(resid, resid, acc);  // count each sample once
    y = y1; Y0 = Y1; dY0 = dY1;
  }
  float term = Y0 - y * y;                          // g(y) = y^2
  if (lane < 16) acc = fmaf(term, term, acc);

  // wave reduce (upper half contributes 0)
  acc = swzadd<1>(acc);
  acc = swzadd<2>(acc);
  acc = swzadd<4>(acc);
  acc = swzadd<8>(acc);
  acc = swzadd<16>(acc);
  if (lane == 0) sPart[wave] = acc;
  __syncthreads();
  if (tid == 0) {
    float s = 0.f;
#pragma unroll
    for (int w = 0; w < 8; ++w) s += sPart[w];
    partials[blockIdx.x] = s;
  }
}

__global__ __launch_bounds__(256) void fbsnn_finalize(
    const float* __restrict__ partials, int n, float* __restrict__ out, float invB) {
  __shared__ float s[256];
  float a = 0.f;
  for (int i = threadIdx.x; i < n; i += 256) a += partials[i];
  s[threadIdx.x] = a;
  __syncthreads();
  for (int d = 128; d > 0; d >>= 1) {
    if ((int)threadIdx.x < d) s[threadIdx.x] += s[threadIdx.x + d];
    __syncthreads();
  }
  if (threadIdx.x == 0) out[0] = s[0] * invB;
}

extern "C" void kernel_launch(void* const* d_in, const int* in_sizes, int n_in,
                              void* d_out, int out_size, void* d_ws, size_t ws_size,
                              hipStream_t stream) {
  const float* dW  = (const float*)d_in[0];
  const float* y0i = (const float*)d_in[1];
  const float* Yw1 = (const float*)d_in[2];
  const float* Yb1 = (const float*)d_in[3];
  const float* Yw2 = (const float*)d_in[4];
  const float* Yb2 = (const float*)d_in[5];
  const float* Yw3 = (const float*)d_in[6];
  const float* Yb3 = (const float*)d_in[7];
  const float* Qw1 = (const float*)d_in[8];
  const float* Qb1 = (const float*)d_in[9];
  const float* Qw2 = (const float*)d_in[10];
  const float* Qb2 = (const float*)d_in[11];
  const float* Qw3 = (const float*)d_in[12];
  const float* Qb3 = (const float*)d_in[13];

  int B = in_sizes[0] / NSTEPS_C;          // dW is [50, B, 1]
  int nblocks = B / 128;                   // 8 waves * 16 samples per block
  float* partials = (float*)d_ws;

  fbsnn_main<<<dim3(nblocks), dim3(256), 0, stream>>>(
      dW, y0i, Yw1, Yb1, Yw2, Yb2, Yw3, Yb3,
      Qw1, Qb1, Qw2, Qb2, Qw3, Qb3, partials, B);
  fbsnn_finalize<<<dim3(1), dim3(256), 0, stream>>>(
      partials, nblocks, (float*)d_out, 1.0f / (float)B);
}
